// RotateQuantizedLinear_9242769622432
// MI455X (gfx1250) — matrix operand order repaired and checked by reading, not compiled, not run on hardware
//
#include <hip/hip_runtime.h>

typedef __bf16 v16bf __attribute__((ext_vector_type(16)));
typedef float  v8f   __attribute__((ext_vector_type(8)));

#define TOK     8192
#define DIM     4096
#define OUTN    4096
#define KROT    8
#define GS      128
#define NGROUPS 32
#define NPAIR   2048   // DIM/2

// ---------- helpers ----------
__device__ __forceinline__ unsigned short f2bf(float f) {
    union { float f; unsigned int u; } v; v.f = f;
    unsigned int u = v.u;
    unsigned int r = u + 0x7FFFu + ((u >> 16) & 1u);   // round-to-nearest-even
    return (unsigned short)(r >> 16);
}

// ---------- 1) cos/sin tables ----------
__global__ void trig_kernel(const float* __restrict__ theta,
                            float* __restrict__ cosT, float* __restrict__ sinT, int n) {
    int i = blockIdx.x * blockDim.x + threadIdx.x;
    if (i < n) {
        float t = theta[i];
        cosT[i] = cosf(t);
        sinT[i] = sinf(t);
    }
}

// ---------- 2) sequential Givens rotations + channel scale -> bf16 ----------
// grid (NGROUPS, TOK/4), block 128. Each block: 4 token rows x one 128-ch group.
__global__ void rotate_kernel(const float* __restrict__ x,
                              const int*   __restrict__ pairs,
                              const float* __restrict__ cosT,
                              const float* __restrict__ sinT,
                              const float* __restrict__ chs,
                              unsigned short* __restrict__ xr) {
    __shared__ float tile[4][GS + 4];
    const int g     = blockIdx.x;
    const int row0  = blockIdx.y * 4;
    const int t     = threadIdx.x;          // 0..127
    const int cbase = g * GS;

    #pragma unroll
    for (int r = 0; r < 4; ++r)
        tile[r][t] = x[(size_t)(row0 + r) * DIM + cbase + t];

    for (int k = 0; k < KROT; ++k) {        // sequential over rotations (scan order)
        __syncthreads();
        if (t < GS / 2) {                   // 64 disjoint pairs per group per pass
            const int i  = pairs[(size_t)k * DIM + cbase + 2 * t];
            const int j  = pairs[(size_t)k * DIM + cbase + 2 * t + 1];
            const int gp = g * (GS / 2) + t;
            const float c = cosT[k * NPAIR + gp];
            const float s = sinT[k * NPAIR + gp];
            #pragma unroll
            for (int r = 0; r < 4; ++r) {
                const float xi = tile[r][i];
                const float xj = tile[r][j];
                tile[r][i] = c * xi - s * xj;
                tile[r][j] = s * xi + c * xj;
            }
        }
    }
    __syncthreads();
    const float cs = chs[cbase + t];
    #pragma unroll
    for (int r = 0; r < 4; ++r)
        xr[(size_t)(row0 + r) * DIM + cbase + t] = f2bf(tile[r][t] * cs);
}

// ---------- 3) dequant int4-in-int32 -> bf16, transposed to [N][K] ----------
// grid (OUTN/32, DIM/32), block (32,8). 32k x 32n tile, LDS transpose.
__global__ void dequant_kernel(const int*   __restrict__ qw,
                               const int*   __restrict__ zeros,
                               const float* __restrict__ scales,
                               unsigned short* __restrict__ wdqT) {
    __shared__ float tile[32][33];
    const int n0 = blockIdx.x * 32;
    const int k0 = blockIdx.y * 32;
    const int tx = threadIdx.x;             // n within tile
    const int ty = threadIdx.y;             // 0..7
    const int g  = k0 >> 7;                 // constant per block (32 | 128)
    const int n  = n0 + tx;
    const float z = (float)zeros[(size_t)g * OUTN + n];
    const float s = scales[(size_t)g * OUTN + n];
    #pragma unroll
    for (int i = 0; i < 4; ++i) {
        const int kl = ty * 4 + i;
        const int q  = qw[(size_t)(k0 + kl) * OUTN + n];
        tile[kl][tx] = ((float)q - z) * s;
    }
    __syncthreads();
    #pragma unroll
    for (int i = 0; i < 4; ++i) {
        const int nl = ty * 4 + i;
        wdqT[(size_t)(n0 + nl) * DIM + k0 + tx] = f2bf(tile[tx][nl]);  // [n][k]
    }
}

// ---------- 4) bf16 WMMA GEMM with async global->LDS double buffering ----------
// C[8192x4096] = A[8192x4096] * B^T (wdqT is N-major [OUTN][DIM])
// Block: 128 threads = 4 waves (2x2), wave tile 64x64 = 4x4 WMMA tiles.
// KT=64: two K=32 WMMA sub-steps per buffered tile -> 32 WMMAs between barriers.
#define BM  128
#define BN  128
#define KT  64
#define AST 72   // LDS row stride in bf16 (144B, 16B aligned)
#define BST 72

union Frag { v16bf v; uint4 q[2]; };

__global__ __launch_bounds__(128)
void gemm_kernel(const unsigned short* __restrict__ A,   // xr   [TOK][DIM]
                 const unsigned short* __restrict__ Bt,  // wdqT [OUTN][DIM]
                 float* __restrict__ C) {
    __shared__ __align__(16) unsigned short As[2][BM * AST];
    __shared__ __align__(16) unsigned short Bs[2][BN * BST];

    const int tid  = threadIdx.x;
    const int wave = tid >> 5;              // 0..3
    const int lane = tid & 31;
    const int half = lane >> 4;
    const int l16  = lane & 15;
    const int wm   = wave >> 1;             // 0..1 : 64-row slab
    const int wn   = wave & 1;              // 0..1 : 64-col slab
    const int bm   = blockIdx.y * BM;
    const int bn   = blockIdx.x * BN;

    const int r0 = tid >> 3;                // 0..15 (16 rows x 8 chunks covers 128x64)
    const int c8 = (tid & 7) * 8;           // bf16 offset within 64-wide K tile

    const unsigned short* Ag = A  + (size_t)(bm + r0) * DIM + c8;
    const unsigned short* Bg = Bt + (size_t)(bn + r0) * DIM + c8;

    // async fill of one double-buffer slot: 8 A rows + 8 B rows per thread (b128 each)
    auto issue_tile = [&](int p, int k0) {
        #pragma unroll
        for (int i = 0; i < 8; ++i) {
            unsigned la = (unsigned)(uintptr_t)&As[p][(r0 + 16 * i) * AST + c8];
            unsigned long long ga =
                (unsigned long long)(uintptr_t)(Ag + (size_t)(16 * i) * DIM + k0);
            asm volatile("global_load_async_to_lds_b128 %0, %1, off"
                         :: "v"(la), "v"(ga) : "memory");
        }
        #pragma unroll
        for (int i = 0; i < 8; ++i) {
            unsigned lb = (unsigned)(uintptr_t)&Bs[p][(r0 + 16 * i) * BST + c8];
            unsigned long long gb =
                (unsigned long long)(uintptr_t)(Bg + (size_t)(16 * i) * DIM + k0);
            asm volatile("global_load_async_to_lds_b128 %0, %1, off"
                         :: "v"(lb), "v"(gb) : "memory");
        }
    };

    v8f acc[4][4];
    #pragma unroll
    for (int tm = 0; tm < 4; ++tm)
        #pragma unroll
        for (int tn = 0; tn < 4; ++tn)
            acc[tm][tn] = (v8f)0.0f;

    issue_tile(0, 0);

    const int NIT = DIM / KT;               // 64
    for (int it = 0; it < NIT; ++it) {
        const int p = it & 1;

        // tile `it` fully resident in buffer p (for all waves)
        asm volatile("s_wait_asynccnt 0x0" ::: "memory");
        __syncthreads();

        // overlap: start filling the other buffer while we compute
        if (it + 1 < NIT) issue_tile(p ^ 1, (it + 1) * KT);

        // ---- two K=32 sub-steps, 16 WMMAs each, frag regs reused
        #pragma unroll
        for (int ks = 0; ks < KT; ks += 32) {
            Frag a[4], b[4];
            #pragma unroll
            for (int tm = 0; tm < 4; ++tm) {
                const int row = wm * 64 + tm * 16 + l16;        // M = l16
                a[tm].q[0] = *(const uint4*)&As[p][row * AST + ks + half * 8];
                a[tm].q[1] = *(const uint4*)&As[p][row * AST + ks + 16 + half * 8];
            }
            #pragma unroll
            for (int tn = 0; tn < 4; ++tn) {
                const int n = wn * 64 + tn * 16 + l16;          // N = l16
                b[tn].q[0] = *(const uint4*)&Bs[p][n * BST + ks + half * 8];
                b[tn].q[1] = *(const uint4*)&Bs[p][n * BST + ks + 16 + half * 8];
            }
            #pragma unroll
            for (int tm = 0; tm < 4; ++tm)
                #pragma unroll
                for (int tn = 0; tn < 4; ++tn)
                    acc[tm][tn] = __builtin_amdgcn_wmma_f32_16x16x32_bf16(
                        false, a[tm].v, false, b[tn].v, (short)0, acc[tm][tn], false, false);
        }

        // all our LDS reads of buffer p done before anyone overwrites it next iter
        asm volatile("s_wait_dscnt 0x0" ::: "memory");
        __syncthreads();
    }

    // ---- store C: VGPR r -> M = r + half*8, N = l16
    #pragma unroll
    for (int tm = 0; tm < 4; ++tm) {
        #pragma unroll
        for (int tn = 0; tn < 4; ++tn) {
            const int row0 = bm + wm * 64 + tm * 16 + half * 8;
            const int col  = bn + wn * 64 + tn * 16 + l16;
            #pragma unroll
            for (int r = 0; r < 8; ++r)
                C[(size_t)(row0 + r) * OUTN + col] = acc[tm][tn][r];
        }
    }
}

// ---------- launch ----------
extern "C" void kernel_launch(void* const* d_in, const int* in_sizes, int n_in,
                              void* d_out, int out_size, void* d_ws, size_t ws_size,
                              hipStream_t stream) {
    const float* x      = (const float*)d_in[0];
    const float* theta  = (const float*)d_in[1];
    const int*   pairs  = (const int*)  d_in[2];
    const float* chs    = (const float*)d_in[3];
    const int*   qw     = (const int*)  d_in[4];
    const float* scales = (const float*)d_in[5];
    const int*   zeros  = (const int*)  d_in[6];
    float*       out    = (float*)d_out;

    char* ws = (char*)d_ws;
    unsigned short* xr   = (unsigned short*)ws;                                   // 64 MB
    unsigned short* wdqT = (unsigned short*)(ws + (size_t)TOK * DIM * 2);         // 32 MB
    float* cosT = (float*)(ws + (size_t)TOK * DIM * 2 + (size_t)DIM * OUTN * 2);  // 64 KB
    float* sinT = cosT + KROT * NPAIR;                                            // 64 KB

    trig_kernel   <<<(KROT * NPAIR + 255) / 256, 256, 0, stream>>>(theta, cosT, sinT, KROT * NPAIR);
    rotate_kernel <<<dim3(NGROUPS, TOK / 4), 128, 0, stream>>>(x, pairs, cosT, sinT, chs, xr);
    dequant_kernel<<<dim3(OUTN / 32, DIM / 32), dim3(32, 8), 0, stream>>>(qw, zeros, scales, wdqT);
    gemm_kernel   <<<dim3(OUTN / BN, TOK / BM), 128, 0, stream>>>(xr, wdqT, out);
}
